// Moeincnet_57380763075298
// MI455X (gfx1250) — matrix operand order repaired and checked by reading, not compiled, not run on hardware
//
#include <hip/hip_runtime.h>
#include <hip/hip_bf16.h>

// ---------------- problem constants ----------------
#define NPTS 65536
#define NE   8
#define FF   236
#define FP   256          // padded F (K and N of expert GEMMs)
#define MM   441
#define MKP  512          // padded K for gating GEMM
#define MNP  448          // padded N for gating GEMM
#define HLN  3
#define W0F  30.0f

typedef __attribute__((ext_vector_type(16))) _Float16 v16h;
typedef __attribute__((ext_vector_type(8)))  _Float16 v8h;
typedef __attribute__((ext_vector_type(8)))  float    v8f;

#if defined(__has_builtin)
#if __has_builtin(__builtin_amdgcn_sched_group_barrier)
#define SCHED_GROUP(mask, size) __builtin_amdgcn_sched_group_barrier(mask, size, 0)
#endif
#endif
#ifndef SCHED_GROUP
#define SCHED_GROUP(mask, size)
#endif

// meta (int) layout inside workspace
// [0..7] counts, [8..16] row offsets, [17..24] cursors, [25..33] block
// offsets (64-row blocks for expert GEMMs), [34] total expert blocks,
// [36..37] dense offsets, [38..39] dense block offsets (32-row), [40] total

// ---------------- init ----------------
__global__ __launch_bounds__(32) void k_init(int* meta) {
  int t = threadIdx.x;
  if (t < 8) { meta[t] = 0; meta[17 + t] = 0; }
  if (t == 0) {
    meta[36] = 0; meta[37] = NPTS;
    meta[38] = 0; meta[39] = NPTS / 32;
    meta[40] = NPTS / 32;
  }
}

// ---------------- f32 -> padded f16 weight conversion ----------------
__global__ __launch_bounds__(256) void k_convert(
    const float* __restrict__ src, _Float16* __restrict__ dst,
    int K, int Nn, int KP2, int NP2, long srcMat, long dstMat, int total) {
  int idx = blockIdx.x * 256 + threadIdx.x;
  if (idx >= total) return;
  int per = KP2 * NP2;
  int m = idx / per;
  int rem = idx - m * per;
  int k = rem / NP2;
  int n = rem - k * NP2;
  float v = (k < K && n < Nn) ? src[(long)m * srcMat + (long)k * Nn + n] : 0.f;
  dst[(long)m * dstMat + (long)k * NP2 + n] = (_Float16)v;
}

// ---------------- gating layer 1 (K=3, VALU) ----------------
__global__ __launch_bounds__(256) void k_gate1(
    const float* __restrict__ x, const float* __restrict__ W1,
    const float* __restrict__ b1, _Float16* __restrict__ g1) {
  int idx = blockIdx.x * 256 + threadIdx.x;           // over NPTS*512
  int n = idx >> 9, m = idx & 511;
  float z = 0.f;
  if (m < MM) {
    z = b1[m] + x[n * 3 + 0] * W1[m] + x[n * 3 + 1] * W1[MM + m] +
        x[n * 3 + 2] * W1[2 * MM + m];
    z = fmaxf(z, 0.f);
  }
  g1[(long)n * MKP + m] = (_Float16)z;
}

// ---------------- generic WMMA GEMM: Out = epi(A @ W + bias) ----------------
// A: f16 [rows][LDA], W: padded f16 [KP][NP] per expert, Out: f16 [rows][LDO]
// 256 threads = 8 waves = 2 row groups x 4 column quarters.  Wave tile =
// (16*ROWT rows) x (NP/4 cols); block tile = (32*ROWT rows) x NP cols.
// sched_group_barrier pins the schedule to (all DS reads) -> (all WMMAs) per
// k-step, forcing distinct operand registers and one dscnt wait per burst.
// EPI: 0=none 1=sin(30z) 2=relu
template <int NP, int KP, int KSLAB, int LDA, int LDO, int EPI, int ROWT>
__global__ __launch_bounds__(256, 2) void gemm_wmma(
    const _Float16* __restrict__ A, const _Float16* __restrict__ Wb,
    long wStride, const float* __restrict__ biasB, int biasStride, int biasN,
    _Float16* __restrict__ Out, const int* __restrict__ offs,
    const int* __restrict__ blk, const int* __restrict__ totalPtr, int nE) {
  constexpr int BM = 32 * ROWT;                 // rows per block
  int bid = blockIdx.x;
  if (bid >= *totalPtr) return;
  int e = 0;
  while (e + 1 < nE && bid >= blk[e + 1]) ++e;
  const int row0   = offs[e] + (bid - blk[e]) * BM;
  const int segEnd = offs[e + 1];
  const _Float16* __restrict__ W = Wb + (long)e * wStride;
  const float* __restrict__ bias = biasB + (long)e * biasStride;

  extern __shared__ char smem[];
  constexpr int S = KSLAB + 8;                  // halves; 16B-aligned rows
  _Float16* At = (_Float16*)smem;               // [BM][S]
  _Float16* Wt = (_Float16*)smem + BM * S;      // [NP][S] (W transposed)

  const int tid = threadIdx.x, wave = tid >> 5, lane = tid & 31;
  const int half = lane >> 4, mr = lane & 15;
  constexpr int NT = NP / 16, NTW = NT / 4;     // 4 (expert) or 7 (gating)
  const int rowG = (wave >> 2) * (16 * ROWT);
  const int colBase = (wave & 3) * (NTW * 16);

  v8f zacc = {0.f, 0.f, 0.f, 0.f, 0.f, 0.f, 0.f, 0.f};
  v8f acc[ROWT][NTW];
#pragma unroll
  for (int rt = 0; rt < ROWT; ++rt)
#pragma unroll
    for (int i = 0; i < NTW; ++i) acc[rt][i] = zacc;

#pragma unroll 1
  for (int k0 = 0; k0 < KP; k0 += KSLAB) {
    if (k0) __syncthreads();
    // stage A tile (clamped gather rows), coalesced b128
    constexpr int ACH = BM * (KSLAB / 8);
#pragma unroll 1
    for (int c = tid; c < ACH; c += 256) {
      int r = c / (KSLAB / 8);
      int kc = (c % (KSLAB / 8)) * 8;
      int gr = row0 + r;
      gr = gr < segEnd ? gr : segEnd - 1;
      *(v8h*)(At + r * S + kc) = *(const v8h*)(A + (long)gr * LDA + k0 + kc);
    }
    // stage W transposed: Wt[n][k]
    constexpr int WCH = NP * KSLAB / 8;
#pragma unroll 1
    for (int c = tid; c < WCH; c += 256) {
      int i = c * 8;
      int kk = i / NP, n = i % NP;
      v8h w = *(const v8h*)(W + (long)(k0 + kk) * NP + n);
#pragma unroll
      for (int j = 0; j < 8; ++j) Wt[(n + j) * S + kk] = w[j];
    }
    __syncthreads();

    for (int ks = 0; ks < KSLAB / 32; ++ks) {
      // A operands first (lane<16 -> K 0..7/16..23, lane>=16 -> +8)
      v16h a[ROWT];
#pragma unroll
      for (int rt = 0; rt < ROWT; ++rt) {
        const _Float16* pa =
            At + (rowG + rt * 16 + mr) * S + ks * 32 + half * 8;
        v8h a0 = *(const v8h*)pa;
        v8h a1 = *(const v8h*)(pa + 16);
        a[rt] = __builtin_shufflevector(a0, a1, 0, 1, 2, 3, 4, 5, 6, 7, 8, 9,
                                        10, 11, 12, 13, 14, 15);
      }
      // all B operands into distinct registers (lane n reads K kb..kb+7, kb+16..kb+23)
      const int kb = ks * 32 + half * 8;
      v16h b[NTW];
#pragma unroll
      for (int nt = 0; nt < NTW; ++nt) {
        const _Float16* pb = Wt + (colBase + nt * 16 + mr) * S + kb;
        v8h b0 = *(const v8h*)pb;
        v8h b1 = *(const v8h*)(pb + 16);
        b[nt] = __builtin_shufflevector(b0, b1, 0, 1, 2, 3, 4, 5, 6, 7, 8, 9,
                                        10, 11, 12, 13, 14, 15);
      }
      // back-to-back WMMAs, no operand register reuse within the step
#pragma unroll
      for (int nt = 0; nt < NTW; ++nt)
#pragma unroll
        for (int rt = 0; rt < ROWT; ++rt)
          acc[rt][nt] = __builtin_amdgcn_wmma_f32_16x16x32_f16(
              false, a[rt], false, b[nt], (short)0, acc[rt][nt], false,
              false);
      // pin the schedule: burst of DS reads, then burst of WMMAs
      SCHED_GROUP(0x100, 2 * ROWT + 2 * NTW);   // DS read group
      SCHED_GROUP(0x008, ROWT * NTW);           // MFMA/WMMA group
    }
  }

  // epilogue: C/D layout -> n = lane%16, m = v + (lane/16)*8
#pragma unroll
  for (int rt = 0; rt < ROWT; ++rt) {
    const int rbase = row0 + rowG + rt * 16 + half * 8;
#pragma unroll
    for (int nt = 0; nt < NTW; ++nt) {
      const int col = colBase + nt * 16 + mr;
      const float bv = (col < biasN) ? bias[col] : 0.f;
#pragma unroll
      for (int v = 0; v < 8; ++v) {
        int r = rbase + v;
        if (r < segEnd) {
          float z = acc[rt][nt][v] + bv;
          if (EPI == 1) z = __sinf(W0F * z);
          else if (EPI == 2) z = fmaxf(z, 0.f);
          Out[(long)r * LDO + col] = (_Float16)z;
        }
      }
    }
  }
}

// ---------------- gating layer 3: logits + argmax + counts ----------------
__global__ __launch_bounds__(256) void k_gate3(
    const _Float16* __restrict__ g2, const float* __restrict__ W3,
    const float* __restrict__ b3, int* __restrict__ topIdx,
    int* __restrict__ counts) {
  int gwave = (blockIdx.x * 256 + threadIdx.x) >> 5;
  int lane = threadIdx.x & 31;
  float acc[NE];
#pragma unroll
  for (int e = 0; e < NE; ++e) acc[e] = 0.f;
  const _Float16* row = g2 + (long)gwave * MNP;
  for (int k = lane; k < MM; k += 32) {
    float g = (float)row[k];
    const float* w = W3 + k * NE;
#pragma unroll
    for (int e = 0; e < NE; ++e) acc[e] += g * w[e];
  }
#pragma unroll
  for (int off = 16; off; off >>= 1)
#pragma unroll
    for (int e = 0; e < NE; ++e) acc[e] += __shfl_xor(acc[e], off, 32);
  if (lane == 0) {
    int best = 0;
    float bv = acc[0] + b3[0];
#pragma unroll
    for (int e = 1; e < NE; ++e) {
      float v = acc[e] + b3[e];
      if (v > bv) { bv = v; best = e; }   // strict '>' keeps first max (top_k)
    }
    topIdx[gwave] = best;
    atomicAdd(&counts[best], 1);
  }
}

// ---------------- scan + KL constant ----------------
__global__ void k_scan(int* meta, float* klOut) {
  int o = 0, b = 0;
  meta[8] = 0; meta[25] = 0;
  for (int e = 0; e < NE; ++e) {
    int c = meta[e];
    o += c;  meta[9 + e]  = o;
    b += (c + 63) / 64;  meta[26 + e] = b;   // 64-row expert blocks
  }
  meta[34] = b;
  // top-1 masked softmax is exactly one-hot -> KL is a data-independent const
  float lu = logf(0.125f);
  float l10 = logf(1e-10f);
  float l1 = logf(1.0f + 1e-10f);
  klOut[0] = 0.125f * (8.0f * lu - 7.0f * l10 - l1);
}

// ---------------- dispatch: perm + gathered coords ----------------
__global__ __launch_bounds__(256) void k_scatter(
    const int* __restrict__ topIdx, const float* __restrict__ x, int* meta,
    int* __restrict__ perm, float* __restrict__ xc) {
  int n = blockIdx.x * 256 + threadIdx.x;
  if (n >= NPTS) return;
  int e = topIdx[n];
  int pos = atomicAdd(&meta[17 + e], 1);
  int r = meta[8 + e] + pos;
  perm[r] = n;
  xc[r * 3 + 0] = x[n * 3 + 0];
  xc[r * 3 + 1] = x[n * 3 + 1];
  xc[r * 3 + 2] = x[n * 3 + 2];
}

// ---------------- SIREN layer 1 (K=3, VALU) on compacted rows ----------------
__global__ __launch_bounds__(256) void k_siren1(
    const float* __restrict__ xc, const float* __restrict__ sW1,
    const float* __restrict__ sb1, const int* __restrict__ offs,
    _Float16* __restrict__ h0) {
  int idx = blockIdx.x * 256 + threadIdx.x;           // over NPTS*FP
  int r = idx >> 8, f = idx & 255;
  float z = 0.f;
  if (f < FF) {
    int e = 0;
    while (e < NE - 1 && r >= offs[e + 1]) ++e;
    const float* w = sW1 + (long)e * 3 * FF;
    float t = sb1[e * FF + f] + xc[r * 3 + 0] * w[f] +
              xc[r * 3 + 1] * w[FF + f] + xc[r * 3 + 2] * w[2 * FF + f];
    z = __sinf(W0F * t);
  }
  h0[(long)r * FP + f] = (_Float16)z;
}

// ---------------- decoder layer 2 (N=3) + scatter to output ----------------
__global__ __launch_bounds__(256) void k_final(
    const _Float16* __restrict__ d1, const float* __restrict__ W2,
    const float* __restrict__ b2, const int* __restrict__ perm,
    float* __restrict__ out) {
  int gwave = (blockIdx.x * 256 + threadIdx.x) >> 5;
  int lane = threadIdx.x & 31;
  float a0 = 0.f, a1 = 0.f, a2 = 0.f;
  const _Float16* row = d1 + (long)gwave * FP;
  for (int k = lane; k < FF; k += 32) {
    float h = (float)row[k];
    a0 += h * W2[k * 3 + 0];
    a1 += h * W2[k * 3 + 1];
    a2 += h * W2[k * 3 + 2];
  }
#pragma unroll
  for (int off = 16; off; off >>= 1) {
    a0 += __shfl_xor(a0, off, 32);
    a1 += __shfl_xor(a1, off, 32);
    a2 += __shfl_xor(a2, off, 32);
  }
  if (lane == 0) {
    int n = perm[gwave];
    out[n * 3 + 0] = a0 + b2[0];
    out[n * 3 + 1] = a1 + b2[1];
    out[n * 3 + 2] = a2 + b2[2];
  }
}

// ---------------- host orchestration ----------------
extern "C" void kernel_launch(void* const* d_in, const int* in_sizes, int n_in,
                              void* d_out, int out_size, void* d_ws,
                              size_t ws_size, hipStream_t stream) {
  (void)in_sizes; (void)n_in; (void)out_size; (void)ws_size;
  const float* x   = (const float*)d_in[0];
  const float* sW1 = (const float*)d_in[1];
  const float* sb1 = (const float*)d_in[2];
  const float* sWh = (const float*)d_in[3];
  const float* sbh = (const float*)d_in[4];
  const float* sWo = (const float*)d_in[5];
  const float* sbo = (const float*)d_in[6];
  const float* mW1 = (const float*)d_in[7];
  const float* mb1 = (const float*)d_in[8];
  const float* mW2 = (const float*)d_in[9];
  const float* mb2 = (const float*)d_in[10];
  const float* mW3 = (const float*)d_in[11];
  const float* mb3 = (const float*)d_in[12];
  const float* dW1 = (const float*)d_in[13];
  const float* db1 = (const float*)d_in[14];
  const float* dW2 = (const float*)d_in[15];
  const float* db2 = (const float*)d_in[16];
  float* out = (float*)d_out;

  // workspace carve (~133 MB)
  char* ws = (char*)d_ws;
  const long SZF = (long)FP * FP;                 // 65536 halves / matrix
  _Float16* WhF = (_Float16*)(ws + 0);            // 24 * 131072 B
  _Float16* WoF = (_Float16*)(ws + 3145728);      //  8 * 131072 B
  _Float16* D1F = (_Float16*)(ws + 4194304);      //  8 * 131072 B
  _Float16* MW2F = (_Float16*)(ws + 5242880);     // 512*448*2 B
  int*   meta   = (int*)(ws + 5701632);
  int*   topIdx = (int*)(ws + 5701888);
  int*   perm   = (int*)(ws + 5964032);
  float* xc     = (float*)(ws + 6226176);
  _Float16* bufA = (_Float16*)(ws + 7012608);     // g1 (N x 512) / h ping
  _Float16* bufB = (_Float16*)(ws + 74121472);    // g2 (N x 448) / h pong

  hipLaunchKernelGGL(k_init, dim3(1), dim3(32), 0, stream, meta);

  // weight conversion to zero-padded f16
  {
    long sm = (long)FF * FF;
    int totWh = 24 * FP * FP;
    hipLaunchKernelGGL(k_convert, dim3((totWh + 255) / 256), dim3(256), 0,
                       stream, sWh, WhF, FF, FF, FP, FP, sm, SZF, totWh);
    int totE = NE * FP * FP;
    hipLaunchKernelGGL(k_convert, dim3((totE + 255) / 256), dim3(256), 0,
                       stream, sWo, WoF, FF, FF, FP, FP, sm, SZF, totE);
    hipLaunchKernelGGL(k_convert, dim3((totE + 255) / 256), dim3(256), 0,
                       stream, dW1, D1F, FF, FF, FP, FP, sm, SZF, totE);
    int totM = MKP * MNP;
    hipLaunchKernelGGL(k_convert, dim3((totM + 255) / 256), dim3(256), 0,
                       stream, mW2, MW2F, MM, MM, MKP, MNP, 0L, 0L, totM);
  }

  // gating
  hipLaunchKernelGGL(k_gate1, dim3(NPTS * 512 / 256), dim3(256), 0, stream,
                     x, mW1, mb1, bufA);
  constexpr int smemGate = (32 + MNP) * (128 + 8) * 2;   // 130560 B
  hipLaunchKernelGGL((gemm_wmma<MNP, MKP, 128, MKP, MNP, 2, 1>),
                     dim3(NPTS / 32), dim3(256), smemGate, stream,
                     bufA, MW2F, 0L, mb2, 0, MM, bufB,
                     meta + 36, meta + 38, meta + 40, 1);
  hipLaunchKernelGGL(k_gate3, dim3(NPTS * 32 / 256), dim3(256), 0, stream,
                     bufB, mW3, mb3, topIdx, meta);
  hipLaunchKernelGGL(k_scan, dim3(1), dim3(1), 0, stream, meta,
                     out + (long)NPTS * 3);
  hipLaunchKernelGGL(k_scatter, dim3(NPTS / 256), dim3(256), 0, stream,
                     topIdx, x, meta, perm, xc);

  // expert pipeline on compacted rows (grid covers worst-case block count)
  hipLaunchKernelGGL(k_siren1, dim3(NPTS * FP / 256), dim3(256), 0, stream,
                     xc, sW1, sb1, meta + 8, bufA);
  constexpr int smemEx = (64 + FP) * (FP + 8) * 2;       // 168960 B
  const int MAXB = NPTS / 64 + NE;
  _Float16* hin = bufA; _Float16* hout = bufB;
  for (int l = 0; l < HLN; ++l) {
    hipLaunchKernelGGL((gemm_wmma<FP, FP, FP, FP, FP, 1, 2>),
                       dim3(MAXB), dim3(256), smemEx, stream,
                       hin, WhF + (long)l * NE * SZF, SZF,
                       sbh + (long)l * NE * FF, FF, FF, hout,
                       meta + 8, meta + 25, meta + 34, NE);
    _Float16* t = hin; hin = hout; hout = t;
  }
  hipLaunchKernelGGL((gemm_wmma<FP, FP, FP, FP, FP, 0, 2>),
                     dim3(MAXB), dim3(256), smemEx, stream,
                     hin, WoF, SZF, sbo, FF, FF, hout,
                     meta + 8, meta + 25, meta + 34, NE);
  hipLaunchKernelGGL((gemm_wmma<FP, FP, FP, FP, FP, 2, 2>),
                     dim3(MAXB), dim3(256), smemEx, stream,
                     hout, D1F, SZF, db1, 0, FF, hin,
                     meta + 8, meta + 25, meta + 34, NE);
  hipLaunchKernelGGL(k_final, dim3(NPTS * 32 / 256), dim3(256), 0, stream,
                     hin, dW2, db2, perm, out);
}
